// LocalizedFiltering_43731357008249
// MI455X (gfx1250) — matrix operand order repaired and checked by reading, not compiled, not run on hardware
//
#include <hip/hip_runtime.h>
#include <hip/hip_bf16.h>

// LocalizedFiltering for MI455X (gfx1250, wave32, WMMA).
// B=4, S=4096, D=2048, DH=1024.
// Compute-bound (275 GFLOP vs ~20us of HBM traffic) -> bf16 WMMA path:
//   v_wmma_f32_16x16x32_bf16, fp32 accumulate, fp32 epilogues.
// Wave tile 64x64 (4x4 accumulators): 32 WMMAs per 32 b128 loads per k-step
// (~43 FLOP/B from cache), block tile 128x256 with 8 waves (2 M x 4 N).
// Workspace: xb bf16 [B,S+1,D] (67MB) + o1b bf16 [B,S+1,DH] (34MB)
//            + deinterleaved bf16 weights (17MB)  ~= 112 MB total.

#define BB 4
#define SS 4096
#define DD 2048
#define DHH 1024

typedef __attribute__((ext_vector_type(16))) __bf16 v16bf;
typedef __attribute__((ext_vector_type(8)))  float  v8f;

union Frag {
  uint4 q[2];
  v16bf v;
};

__device__ __forceinline__ unsigned short f32_to_bf16(float f) {
  unsigned int u = __float_as_uint(f);
  u += 0x7FFFu + ((u >> 16) & 1u);   // round-to-nearest-even
  return (unsigned short)(u >> 16);
}

// ---------------------------------------------------------------------------
// Shared WMMA mainloop for both conv-GEMM stages.
// Computes acc[4][4] += sum_tap A(shifted rows) * W_tap for a 64x64 wave tile.
//   Ab  : bf16 activation matrix, rows of length K, row aRow0+tl holds
//         x_cat[b, tl] of the cache-padded sequence.
//   W0/W1 : bf16 weights [N][K] row-major, tap 0 / tap 1.
// Output token tl consumes rows tl (tap0) and tl+1 (tap1).
// Fragment addressing follows ISA 7.12.2 16-bit A/B VGPR layouts:
//   A: lane(m=L&15,h=L>>4) holds A[m, kk+8h .. +7] and A[m, kk+16+8h .. +7]
//   B: lane(n=L&15,h=L>>4) holds W[n, kk+8h .. +7] and W[n, kk+16+8h .. +7]
// ---------------------------------------------------------------------------
template <int K>
__device__ __forceinline__ void conv_mainloop(
    const unsigned short* __restrict__ Ab, size_t aRow0,
    const unsigned short* __restrict__ W0,
    const unsigned short* __restrict__ W1,
    int tl0, int n0, v8f acc[4][4]) {
  const int lane = threadIdx.x & 31;
  const int mrow = lane & 15;
  const int h    = lane >> 4;

  const unsigned short* aBase  = Ab + (aRow0 + (size_t)(tl0 + mrow)) * K + h * 8;
  const unsigned short* bBase0 = W0 + (size_t)(n0 + mrow) * K + h * 8;
  const unsigned short* bBase1 = W1 + (size_t)(n0 + mrow) * K + h * 8;

  for (int kk = 0; kk < K; kk += 32) {
#pragma unroll
    for (int tap = 0; tap < 2; ++tap) {
      const unsigned short* bb = tap ? bBase1 : bBase0;
      Frag bf[4];
#pragma unroll
      for (int ni = 0; ni < 4; ++ni) {
        const unsigned short* p = bb + (size_t)(ni * 16) * K + kk;
        bf[ni].q[0] = *reinterpret_cast<const uint4*>(p);
        bf[ni].q[1] = *reinterpret_cast<const uint4*>(p + 16);
      }
#pragma unroll
      for (int mi = 0; mi < 4; ++mi) {
        const unsigned short* p = aBase + (size_t)(mi * 16 + tap) * K + kk;
        Frag af;
        af.q[0] = *reinterpret_cast<const uint4*>(p);
        af.q[1] = *reinterpret_cast<const uint4*>(p + 16);
#pragma unroll
        for (int ni = 0; ni < 4; ++ni) {
          acc[mi][ni] = __builtin_amdgcn_wmma_f32_16x16x32_bf16(
              false, af.v, false, bf[ni].v, (short)0, acc[mi][ni], false, false);
        }
      }
    }
  }
}

// ---------------------------------------------------------------------------
// GEMM1: o1[b,t,e] = sum_d w1_0[e,d]*x_cat[t,d] + w1_1[e,d]*x_cat[t+1,d] + b1[e]
// Writes bf16 into o1b row t+1 (o1_cat), and fp32 lf2 output at t==S-1.
// Grid: (B*S/128, DH/256), 256 threads (8 waves, 2x4).
// ---------------------------------------------------------------------------
__global__ __launch_bounds__(256) void gemm1_kernel(
    const unsigned short* __restrict__ xb,
    const unsigned short* __restrict__ w1_0,
    const unsigned short* __restrict__ w1_1,
    const float* __restrict__ b1,
    unsigned short* __restrict__ o1b,
    float* __restrict__ lf2out) {
  const int gBase  = blockIdx.x * 128;
  const int b      = gBase / SS;
  const int tlBase = gBase - b * SS;
  const int mw = (threadIdx.x >> 5) & 1;
  const int nw = threadIdx.x >> 6;
  const int tl0 = tlBase + mw * 64;
  const int n0  = blockIdx.y * 256 + nw * 64;

  v8f acc[4][4];
#pragma unroll
  for (int mi = 0; mi < 4; ++mi)
#pragma unroll
    for (int ni = 0; ni < 4; ++ni)
#pragma unroll
      for (int i = 0; i < 8; ++i) acc[mi][ni][i] = 0.0f;

  conv_mainloop<DD>(xb, (size_t)b * (SS + 1), w1_0, w1_1, tl0, n0, acc);

  const int lane = threadIdx.x & 31;
  const int nlo  = lane & 15;
  const int h    = lane >> 4;
#pragma unroll
  for (int mi = 0; mi < 4; ++mi) {
#pragma unroll
    for (int ni = 0; ni < 4; ++ni) {
      const int n = n0 + ni * 16 + nlo;
      const float bias = b1[n];
#pragma unroll
      for (int v = 0; v < 8; ++v) {
        const int tl = tl0 + mi * 16 + v + 8 * h;      // C/D layout: row = v + 8h
        const float val = acc[mi][ni][v] + bias;
        o1b[((size_t)b * (SS + 1) + tl + 1) * DHH + n] = f32_to_bf16(val);
        if (tl == SS - 1) lf2out[(size_t)b * DHH + n] = val;
      }
    }
  }
}

// ---------------------------------------------------------------------------
// GEMM2: y[b,t,d] = sum_e w2_0[d,e]*o1_cat[t,e] + w2_1[d,e]*o1_cat[t+1,e]
//                   + b2[d] + x[b,t,d]      (residual fused)
// Writes fp32 y into d_out (RMSNorm pass normalizes in place afterwards).
// Grid: (B*S/128, D/256).
// ---------------------------------------------------------------------------
__global__ __launch_bounds__(256) void gemm2_kernel(
    const unsigned short* __restrict__ o1b,
    const unsigned short* __restrict__ w2_0,
    const unsigned short* __restrict__ w2_1,
    const float* __restrict__ b2,
    const float* __restrict__ x,
    float* __restrict__ yout) {
  const int gBase  = blockIdx.x * 128;
  const int b      = gBase / SS;
  const int tlBase = gBase - b * SS;
  const int mw = (threadIdx.x >> 5) & 1;
  const int nw = threadIdx.x >> 6;
  const int tl0 = tlBase + mw * 64;
  const int n0  = blockIdx.y * 256 + nw * 64;

  v8f acc[4][4];
#pragma unroll
  for (int mi = 0; mi < 4; ++mi)
#pragma unroll
    for (int ni = 0; ni < 4; ++ni)
#pragma unroll
      for (int i = 0; i < 8; ++i) acc[mi][ni][i] = 0.0f;

  conv_mainloop<DHH>(o1b, (size_t)b * (SS + 1), w2_0, w2_1, tl0, n0, acc);

  const int lane = threadIdx.x & 31;
  const int nlo  = lane & 15;
  const int h    = lane >> 4;
#pragma unroll
  for (int mi = 0; mi < 4; ++mi) {
#pragma unroll
    for (int ni = 0; ni < 4; ++ni) {
      const int n = n0 + ni * 16 + nlo;
      const float bias = b2[n];
#pragma unroll
      for (int v = 0; v < 8; ++v) {
        const int tl = tl0 + mi * 16 + v + 8 * h;
        const size_t g = (size_t)b * SS + tl;
        yout[g * DD + n] = acc[mi][ni][v] + bias + x[g * DD + n];
      }
    }
  }
}

// ---------------------------------------------------------------------------
// In-place RMSNorm over the channel dim: one block per token row of 2048.
// ---------------------------------------------------------------------------
__global__ __launch_bounds__(256) void rmsnorm_kernel(
    float* __restrict__ y, const float* __restrict__ lnw) {
  __shared__ float red[256];
  float* p = y + (size_t)blockIdx.x * DD;
  float vals[8];
  float ss = 0.0f;
#pragma unroll
  for (int j = 0; j < 8; ++j) {
    vals[j] = p[threadIdx.x + j * 256];
    ss += vals[j] * vals[j];
  }
  red[threadIdx.x] = ss;
  __syncthreads();
  for (int s = 128; s > 0; s >>= 1) {
    if (threadIdx.x < s) red[threadIdx.x] += red[threadIdx.x + s];
    __syncthreads();
  }
  const float inv = rsqrtf(red[0] / (float)DD + 1e-6f);
#pragma unroll
  for (int j = 0; j < 8; ++j) {
    const int c = threadIdx.x + j * 256;
    p[c] = vals[j] * inv * lnw[c];
  }
}

// ---------------------------------------------------------------------------
// Prep passes (bandwidth-bound, trivially small vs GEMM time).
// ---------------------------------------------------------------------------

// xb[b, r, d] = bf16( r==0 ? lf1_cache[b,d] : x[b, r-1, d] )   (x_cat)
__global__ void prep_xb_kernel(const float* __restrict__ inp,
                               const float* __restrict__ lf1c,
                               unsigned short* __restrict__ xb) {
  const size_t idx = (size_t)blockIdx.x * 256 + threadIdx.x;
  const size_t total = (size_t)BB * (SS + 1) * DD;
  if (idx >= total) return;
  const int d = (int)(idx % DD);
  const size_t rb = idx / DD;
  const int r = (int)(rb % (SS + 1));
  const int b = (int)(rb / (SS + 1));
  const float v = (r == 0) ? lf1c[(size_t)b * DD + d]
                           : inp[((size_t)b * SS + (r - 1)) * DD + d];
  xb[idx] = f32_to_bf16(v);
}

// o1b[b, 0, e] = bf16(lf2_cache[b,e])   (rows 1..S filled by gemm1)
__global__ void prep_o1row0_kernel(const float* __restrict__ lf2c,
                                   unsigned short* __restrict__ o1b) {
  const int idx = blockIdx.x * 256 + threadIdx.x;
  if (idx >= BB * DHH) return;
  const int b = idx / DHH;
  const int e = idx % DHH;
  o1b[(size_t)b * (SS + 1) * DHH + e] = f32_to_bf16(lf2c[idx]);
}

// Deinterleave taps: w[(n*K + k)*2 + tap] -> wtap[n*K + k] as bf16.
__global__ void prep_w_kernel(const float* __restrict__ w,
                              unsigned short* __restrict__ w0,
                              unsigned short* __restrict__ w1, int count) {
  const int idx = blockIdx.x * 256 + threadIdx.x;
  if (idx >= count) return;
  w0[idx] = f32_to_bf16(w[2 * (size_t)idx]);
  w1[idx] = f32_to_bf16(w[2 * (size_t)idx + 1]);
}

// lf1 output = x[b, S-1, :] fp32
__global__ void copy_lf1_kernel(const float* __restrict__ inp,
                                float* __restrict__ lf1out) {
  const int idx = blockIdx.x * 256 + threadIdx.x;
  if (idx >= BB * DD) return;
  const int b = idx / DD;
  const int d = idx % DD;
  lf1out[idx] = inp[((size_t)b * SS + (SS - 1)) * DD + d];
}

// ---------------------------------------------------------------------------
extern "C" void kernel_launch(void* const* d_in, const int* in_sizes, int n_in,
                              void* d_out, int out_size, void* d_ws,
                              size_t ws_size, hipStream_t stream) {
  const float* inp  = (const float*)d_in[0];
  const float* lf1c = (const float*)d_in[1];
  const float* lf2c = (const float*)d_in[2];
  const float* w1   = (const float*)d_in[3];
  const float* b1   = (const float*)d_in[4];
  const float* w2   = (const float*)d_in[5];
  const float* b2   = (const float*)d_in[6];
  const float* lnw  = (const float*)d_in[7];

  float* out    = (float*)d_out;                       // lf_output [B,S,D]
  float* lf1out = out + (size_t)BB * SS * DD;          // lf1 [B,D]
  float* lf2out = lf1out + (size_t)BB * DD;            // lf2 [B,DH]

  unsigned short* xb   = (unsigned short*)d_ws;               // B*(S+1)*D
  unsigned short* o1b  = xb   + (size_t)BB * (SS + 1) * DD;   // B*(S+1)*DH
  unsigned short* w1_0 = o1b  + (size_t)BB * (SS + 1) * DHH;  // DH*D
  unsigned short* w1_1 = w1_0 + (size_t)DHH * DD;
  unsigned short* w2_0 = w1_1 + (size_t)DHH * DD;             // D*DH
  unsigned short* w2_1 = w2_0 + (size_t)DD * DHH;

  {
    const size_t total = (size_t)BB * (SS + 1) * DD;
    prep_xb_kernel<<<(int)((total + 255) / 256), 256, 0, stream>>>(inp, lf1c, xb);
  }
  prep_o1row0_kernel<<<(BB * DHH + 255) / 256, 256, 0, stream>>>(lf2c, o1b);
  prep_w_kernel<<<(DHH * DD + 255) / 256, 256, 0, stream>>>(w1, w1_0, w1_1, DHH * DD);
  prep_w_kernel<<<(DD * DHH + 255) / 256, 256, 0, stream>>>(w2, w2_0, w2_1, DD * DHH);
  copy_lf1_kernel<<<(BB * DD + 255) / 256, 256, 0, stream>>>(inp, lf1out);

  gemm1_kernel<<<dim3((BB * SS) / 128, DHH / 256), 256, 0, stream>>>(
      xb, w1_0, w1_1, b1, o1b, lf2out);
  gemm2_kernel<<<dim3((BB * SS) / 128, DD / 256), 256, 0, stream>>>(
      o1b, w2_0, w2_1, b2, inp, out);
  rmsnorm_kernel<<<BB * SS, 256, 0, stream>>>(out, lnw);
}
